// RSSM_49667001811294
// MI455X (gfx1250) — matrix operand order repaired and checked by reading, not compiled, not run on hardware
//
#include <hip/hip_runtime.h>
#include <math.h>

#define Bn   1024
#define Tn   64
#define DET  1024
#define STO  32
#define EMB  1024
#define ACTD 32

typedef _Float16 h16;
typedef __attribute__((ext_vector_type(16))) _Float16 v16h;
typedef __attribute__((ext_vector_type(8)))  float    v8f;

struct alignas(16) B16 { unsigned a, b, c, d; };

#if __has_builtin(__builtin_amdgcn_global_load_async_to_lds_b128)
#define HAS_ASYNC_LDS 1
#else
#define HAS_ASYNC_LDS 0
#endif

// exact param types per hipcc diagnostic: v4i* in AS(1) / AS(3)
typedef int i4vec __attribute__((vector_size(16)));
typedef __attribute__((address_space(1))) i4vec* gp_i4;
typedef __attribute__((address_space(3))) i4vec* lp_i4;

// 16-byte global -> LDS copy. Async (ASYNCcnt) when the gfx1250 builtin is
// available; otherwise plain load+store (compiler inserts waits at barriers).
__device__ __forceinline__ void copy16_g2l(const h16* g, h16* l) {
#if HAS_ASYNC_LDS
  __builtin_amdgcn_global_load_async_to_lds_b128(
      (gp_i4)(void*)(h16*)g, (lp_i4)(void*)l, 0, 0);
#else
  *reinterpret_cast<B16*>(l) = *reinterpret_cast<const B16*>(g);
#endif
}

__device__ __forceinline__ void wait_g2l() {
#if HAS_ASYNC_LDS
#if __has_builtin(__builtin_amdgcn_s_wait_asynccnt)
  __builtin_amdgcn_s_wait_asynccnt(0);
#else
  asm volatile("s_wait_asynccnt 0x0" ::: "memory");
#endif
#endif
}

__device__ __forceinline__ float sigmoid_f(float x) { return 1.f / (1.f + expf(-x)); }
__device__ __forceinline__ float softplus_f(float x) { return x > 20.f ? x : log1pf(expf(x)); }
__device__ __forceinline__ float elu_f(float x) { return x > 0.f ? x : (expf(x) - 1.f); }

// ---------------------------------------------------------------------------
// WMMA GEMM:  C[M x N] = act( A[M x K] * W[N x K]^T + bias[N] )
// A, W row-major f16.  act==0: f32 -> Cf.  act==1: ELU -> f16 Ch.
// M % 128 == 0, K % 32 == 0, N arbitrary (bounds-guarded; grid covers ceil(N/128)).
// Block 256 threads = 8 wave32 (4 over M x 2 over N); block tile 128x128;
// wave tile 32x64 = 2x4 WMMA 16x16x32 tiles. Double-buffered LDS, async loads.
// ---------------------------------------------------------------------------
__global__ __launch_bounds__(256) void gemm_wmma_kernel(
    const h16* __restrict__ A, const h16* __restrict__ W,
    const float* __restrict__ bias,
    float* __restrict__ Cf, h16* __restrict__ Ch,
    int M, int N, int K, int act)
{
  // rows padded to 40 halfs (80B) -> conflict-free ds_load_b128 fragments
  __shared__ __align__(16) h16 ldsA[2][128 * 40];
  __shared__ __align__(16) h16 ldsW[2][128 * 40];

  const int tid  = threadIdx.x;
  const int lane = tid & 31;
  const int wave = tid >> 5;
  const int l16  = lane & 15;
  const int hi   = lane >> 4;     // lane half: 0 or 1
  const int wy   = wave & 3;      // wave row (M), 0..3 -> 32 rows each
  const int wx   = wave >> 2;     // wave col (N), 0..1 -> 64 cols each

  const int n0 = blockIdx.x * 128;
  const int m0 = blockIdx.y * 128;

  v8f acc[2][4] = {};

  // tile staging: 128 rows x 32 halfs each for A and W; 2 threads/row,
  // 16 halfs (2 x b128) per thread per tile.
  const int trow = tid >> 1;           // 0..127
  const int tseg = (tid & 1) * 16;     // 0 or 16 (halfs)

  const h16* gA = A + (size_t)(m0 + trow) * K + tseg;
  const h16* gW = W + (size_t)(n0 + trow) * K + tseg;
  const bool wok = (n0 + trow) < N;    // W rows beyond N: skip (dead columns)

  const int nk = K / 32;

  {  // prologue: chunk 0 -> buffer 0
    h16* dA = &ldsA[0][trow * 40 + tseg];
    copy16_g2l(gA,     dA);
    copy16_g2l(gA + 8, dA + 8);
    if (wok) {
      h16* dW = &ldsW[0][trow * 40 + tseg];
      copy16_g2l(gW,     dW);
      copy16_g2l(gW + 8, dW + 8);
    }
  }

  for (int i = 0; i < nk; ++i) {
    const int buf = i & 1;
    wait_g2l();        // buffer `buf` resident
    __syncthreads();   // all waves done writing buf / done reading buf^1
    if (i + 1 < nk) {  // prefetch next chunk into the other buffer
      const h16* sA = gA + (size_t)(i + 1) * 32;
      const h16* sW = gW + (size_t)(i + 1) * 32;
      h16* dA = &ldsA[buf ^ 1][trow * 40 + tseg];
      copy16_g2l(sA,     dA);
      copy16_g2l(sA + 8, dA + 8);
      if (wok) {
        h16* dW = &ldsW[buf ^ 1][trow * 40 + tseg];
        copy16_g2l(sW,     dW);
        copy16_g2l(sW + 8, dW + 8);
      }
    }

    // Fragments per ISA 7.12.2 (16-bit A 16x32, B 32x16, wave32):
    //  A: lane(l16,hi) = row m=l16, K chunks {8*hi..+7, 16+8*hi..+7}
    //  B: lane(l16,hi) = col n=l16, K chunks {8*hi..+7, 16+8*hi..+7}
    v16h afrag[2], bfrag[4];
#pragma unroll
    for (int mt = 0; mt < 2; ++mt) {
      const int row = wy * 32 + mt * 16 + l16;
      B16* ap = reinterpret_cast<B16*>(&afrag[mt]);
      ap[0] = *reinterpret_cast<const B16*>(&ldsA[buf][row * 40 + hi * 8]);
      ap[1] = *reinterpret_cast<const B16*>(&ldsA[buf][row * 40 + 16 + hi * 8]);
    }
#pragma unroll
    for (int nt = 0; nt < 4; ++nt) {
      const int row = wx * 64 + nt * 16 + l16;
      B16* bp = reinterpret_cast<B16*>(&bfrag[nt]);
      const B16* s = reinterpret_cast<const B16*>(&ldsW[buf][row * 40 + hi * 8]);
      bp[0] = s[0];
      bp[1] = s[2];
    }
#pragma unroll
    for (int mt = 0; mt < 2; ++mt)
#pragma unroll
      for (int nt = 0; nt < 4; ++nt)
        acc[mt][nt] = __builtin_amdgcn_wmma_f32_16x16x32_f16(
            false, afrag[mt], false, bfrag[nt], (short)0, acc[mt][nt], false, false);
  }

  // Epilogue: acc element r -> (m = base + r + 8*hi, n = base + l16)
#pragma unroll
  for (int nt = 0; nt < 4; ++nt) {
    const int n = n0 + wx * 64 + nt * 16 + l16;
    if (n >= N) continue;
    const float bv = bias ? bias[n] : 0.f;
#pragma unroll
    for (int mt = 0; mt < 2; ++mt) {
      const int mbase = m0 + wy * 32 + mt * 16 + hi * 8;
#pragma unroll
      for (int r = 0; r < 8; ++r) {
        float v = acc[mt][nt][r] + bv;
        const size_t idx = (size_t)(mbase + r) * N + n;
        if (act == 1) {
          Ch[idx] = (h16)elu_f(v);
        } else {
          Cf[idx] = v;
        }
      }
    }
  }
}

// ---------------------------------------------------------------------------
// Elementwise helpers
// ---------------------------------------------------------------------------
__global__ void cvt_f32_to_f16_kernel(const float* __restrict__ src, h16* __restrict__ dst, int n) {
  int i = blockIdx.x * blockDim.x + threadIdx.x;
  if (i < n) dst[i] = (h16)src[i];
}

__global__ void copy_f32_kernel(const float* __restrict__ src, float* __restrict__ dst, int n) {
  int i = blockIdx.x * blockDim.x + threadIdx.x;
  if (i < n) dst[i] = src[i];
}

// cat_sa[b][j] = j<32 ? stoch[b][j] : actions[b][t][j-32]   (f16)
__global__ void concat_sa_kernel(const float* __restrict__ stoch, const float* __restrict__ actions,
                                 h16* __restrict__ cat, int t, int n) {
  int i = blockIdx.x * blockDim.x + threadIdx.x;
  if (i >= n) return;
  int b = i >> 6, j = i & 63;
  float v = (j < STO) ? stoch[b * STO + j]
                      : actions[((size_t)b * Tn + t) * ACTD + (j - STO)];
  cat[i] = (h16)v;
}

// cat_de[b][j] = j<1024 ? deter[b][j] : embeds[b][t][j-1024]   (f16)
__global__ void concat_de_kernel(const h16* __restrict__ deter_h, const float* __restrict__ embeds,
                                 h16* __restrict__ cat, int t, int n) {
  int i = blockIdx.x * blockDim.x + threadIdx.x;
  if (i >= n) return;
  int b = i >> 11, j = i & 2047;
  cat[i] = (j < DET) ? deter_h[b * DET + j]
                     : (h16)embeds[((size_t)b * Tn + t) * EMB + (j - DET)];
}

// torch-style GRU gating (gi = x@Wih^T+bih, gh = h@Whh^T+bhh already computed)
__global__ void gru_kernel(const float* __restrict__ gi, const float* __restrict__ gh,
                           float* __restrict__ deter_f, h16* __restrict__ deter_h,
                           float* __restrict__ out_deter, int t, int n) {
  int i = blockIdx.x * blockDim.x + threadIdx.x;
  if (i >= n) return;
  int b = i >> 10, d = i & 1023;
  size_t base = (size_t)b * (3 * DET);
  float r  = sigmoid_f(gi[base + d]           + gh[base + d]);
  float z  = sigmoid_f(gi[base + DET + d]     + gh[base + DET + d]);
  float nn = tanhf(   gi[base + 2 * DET + d]  + r * gh[base + 2 * DET + d]);
  float nd = (1.f - z) * nn + z * deter_f[i];
  deter_f[i] = nd;
  deter_h[i] = (h16)nd;
  out_deter[((size_t)b * Tn + t) * DET + d] = nd;
}

// split mean/logstd, std = softplus+0.1, stoch = mean + noise*std
__global__ void dist_out_kernel(const float* __restrict__ pr, const float* __restrict__ noise,
                                float* __restrict__ o_mean, float* __restrict__ o_std,
                                float* __restrict__ o_stoch, float* __restrict__ stoch_state,
                                int t, int n) {
  int i = blockIdx.x * blockDim.x + threadIdx.x;
  if (i >= n) return;
  int b = i >> 5, s = i & 31;
  float mean   = pr[(size_t)b * 64 + s];
  float logstd = pr[(size_t)b * 64 + 32 + s];
  float sd = softplus_f(logstd) + 0.1f;
  float st = mean + noise[((size_t)b * Tn + t) * STO + s] * sd;
  size_t o = ((size_t)b * Tn + t) * STO + s;
  o_mean[o]  = mean;
  o_std[o]   = sd;
  o_stoch[o] = st;
  if (stoch_state) stoch_state[i] = st;
}

// ---------------------------------------------------------------------------
extern "C" void kernel_launch(void* const* d_in, const int* in_sizes, int n_in,
                              void* d_out, int out_size, void* d_ws, size_t ws_size,
                              hipStream_t stream) {
  (void)in_sizes; (void)n_in; (void)out_size; (void)ws_size;
  const float* embeds      = (const float*)d_in[0];
  const float* actions     = (const float*)d_in[1];
  const float* deter0      = (const float*)d_in[2];
  const float* stoch0      = (const float*)d_in[3];
  const float* noise_prior = (const float*)d_in[4];
  const float* noise_post  = (const float*)d_in[5];
  const float* asp_W = (const float*)d_in[6];
  const float* asp_b = (const float*)d_in[7];
  const float* w_ih  = (const float*)d_in[8];
  const float* w_hh  = (const float*)d_in[9];
  const float* b_ih  = (const float*)d_in[10];
  const float* b_hh  = (const float*)d_in[11];
  const float* rp_W1 = (const float*)d_in[12];
  const float* rp_b1 = (const float*)d_in[13];
  const float* rp_W2 = (const float*)d_in[14];
  const float* rp_b2 = (const float*)d_in[15];
  const float* po_W1 = (const float*)d_in[16];
  const float* po_b1 = (const float*)d_in[17];
  const float* po_W2 = (const float*)d_in[18];
  const float* po_b2 = (const float*)d_in[19];

  float* out = (float*)d_out;
  const size_t S1 = (size_t)Bn * Tn * STO;
  float* out_qmean  = out;
  float* out_qstd   = out + S1;
  float* out_qstoch = out + 2 * S1;
  float* out_pmean  = out + 3 * S1;
  float* out_pstd   = out + 4 * S1;
  float* out_pstoch = out + 5 * S1;
  float* out_deter  = out + 6 * S1;

  char* ws = (char*)d_ws;
  size_t off = 0;
  auto carve = [&](size_t bytes) -> char* {
    char* p = ws + off;
    off += (bytes + 255) & ~(size_t)255;
    return p;
  };
  h16* aspW_h  = (h16*)carve((size_t)DET * 64 * 2);
  h16* wih_h   = (h16*)carve((size_t)3 * DET * DET * 2);
  h16* whh_h   = (h16*)carve((size_t)3 * DET * DET * 2);
  h16* rpW1_h  = (h16*)carve((size_t)DET * DET * 2);
  h16* rpW2_h  = (h16*)carve((size_t)64 * DET * 2);
  h16* poW1_h  = (h16*)carve((size_t)DET * 2048 * 2);
  h16* poW2_h  = (h16*)carve((size_t)64 * DET * 2);
  h16* cat_sa_h = (h16*)carve((size_t)Bn * 64 * 2);
  h16* h_h      = (h16*)carve((size_t)Bn * DET * 2);
  h16* deter_h  = (h16*)carve((size_t)Bn * DET * 2);
  h16* x1_h     = (h16*)carve((size_t)Bn * DET * 2);
  h16* cat_de_h = (h16*)carve((size_t)Bn * 2048 * 2);
  h16* x2_h     = (h16*)carve((size_t)Bn * DET * 2);
  float* gi_f    = (float*)carve((size_t)Bn * 3 * DET * 4);
  float* gh_f    = (float*)carve((size_t)Bn * 3 * DET * 4);
  float* deter_f = (float*)carve((size_t)Bn * DET * 4);
  float* stoch_f = (float*)carve((size_t)Bn * STO * 4);
  float* pr_f    = (float*)carve((size_t)Bn * 64 * 4);
  float* po_f    = (float*)carve((size_t)Bn * 64 * 4);

  const int TPB = 256;
  auto g1 = [&](size_t n) { return dim3((unsigned)((n + TPB - 1) / TPB)); };
  auto cvt = [&](const float* s, h16* d, size_t n) {
    cvt_f32_to_f16_kernel<<<g1(n), TPB, 0, stream>>>(s, d, (int)n);
  };
  auto gemm = [&](const h16* A, const h16* W, const float* bias,
                  float* Cf, h16* Ch, int M, int N, int K, int act) {
    dim3 grid((N + 127) / 128, M / 128);
    gemm_wmma_kernel<<<grid, TPB, 0, stream>>>(A, W, bias, Cf, Ch, M, N, K, act);
  };

  // per-launch (deterministic): weights -> f16, state init from inputs
  cvt(asp_W, aspW_h, (size_t)DET * 64);
  cvt(w_ih,  wih_h,  (size_t)3 * DET * DET);
  cvt(w_hh,  whh_h,  (size_t)3 * DET * DET);
  cvt(rp_W1, rpW1_h, (size_t)DET * DET);
  cvt(rp_W2, rpW2_h, (size_t)64 * DET);
  cvt(po_W1, poW1_h, (size_t)DET * 2048);
  cvt(po_W2, poW2_h, (size_t)64 * DET);
  copy_f32_kernel<<<g1((size_t)Bn * DET), TPB, 0, stream>>>(deter0, deter_f, Bn * DET);
  cvt(deter0, deter_h, (size_t)Bn * DET);
  copy_f32_kernel<<<g1((size_t)Bn * STO), TPB, 0, stream>>>(stoch0, stoch_f, Bn * STO);

  for (int t = 0; t < Tn; ++t) {
    // prior: h = elu([stoch, a] @ aspW^T + asp_b)
    concat_sa_kernel<<<g1((size_t)Bn * 64), TPB, 0, stream>>>(stoch_f, actions, cat_sa_h, t, Bn * 64);
    gemm(cat_sa_h, aspW_h, asp_b, nullptr, h_h, Bn, DET, 64, 1);
    // GRU gates
    gemm(h_h,     wih_h, b_ih, gi_f, nullptr, Bn, 3 * DET, DET, 0);
    gemm(deter_h, whh_h, b_hh, gh_f, nullptr, Bn, 3 * DET, DET, 0);
    gru_kernel<<<g1((size_t)Bn * DET), TPB, 0, stream>>>(gi_f, gh_f, deter_f, deter_h,
                                                         out_deter, t, Bn * DET);
    // prior projector
    gemm(deter_h, rpW1_h, rp_b1, nullptr, x1_h, Bn, DET, DET, 1);
    gemm(x1_h,    rpW2_h, rp_b2, pr_f, nullptr, Bn, 64, DET, 0);
    dist_out_kernel<<<g1((size_t)Bn * STO), TPB, 0, stream>>>(
        pr_f, noise_prior, out_pmean, out_pstd, out_pstoch, nullptr, t, Bn * STO);
    // posterior
    concat_de_kernel<<<g1((size_t)Bn * 2048), TPB, 0, stream>>>(deter_h, embeds, cat_de_h, t, Bn * 2048);
    gemm(cat_de_h, poW1_h, po_b1, nullptr, x2_h, Bn, DET, 2048, 1);
    gemm(x2_h,     poW2_h, po_b2, po_f, nullptr, Bn, 64, DET, 0);
    dist_out_kernel<<<g1((size_t)Bn * STO), TPB, 0, stream>>>(
        po_f, noise_post, out_qmean, out_qstd, out_qstoch, stoch_f, t, Bn * STO);
  }
}
